// RoIProcessingNetwork_87110526698175
// MI455X (gfx1250) — matrix operand order repaired and checked by reading, not compiled, not run on hardware
//
#include <hip/hip_runtime.h>
#include <hip/hip_bf16.h>
#include <math.h>

typedef __bf16 bf16t;
typedef bf16t v16bf __attribute__((ext_vector_type(16)));
typedef bf16t v8bf  __attribute__((ext_vector_type(8)));
typedef float v8f   __attribute__((ext_vector_type(8)));
typedef unsigned int u32x4 __attribute__((ext_vector_type(4)));
typedef int i32x4 __attribute__((ext_vector_type(4)));
typedef int i32x8 __attribute__((ext_vector_type(8)));

#define N_PROP 1000
#define MROWS  1024          // padded M for GEMM (128-multiple)
#define C_CH   256
#define K1     12544         // 256*7*7
#define CLIN   1024
#define N_IMG  2
#define IMTOP  100
#define ASTR   40            // LDS A row stride (elems): 32 data + 8 pad = 80B (bank spread)
#define BSTR   72            // LDS B row stride (elems): 64 data + 8 pad = 144B

// ---- CDNA5 TDM availability / arity discrimination -------------------------
#if defined(__has_builtin)
# if __has_builtin(__builtin_amdgcn_tensor_load_to_lds)
#  define HAVE_TDM 1
# endif
#endif
#ifndef HAVE_TDM
# define HAVE_TDM 0
#endif
#if defined(__has_include)
# if __has_include(<hip/amd_detail/amd_gfx1250_TDM.h>)
#  define TDM_SIX_ARGS 1
# else
#  define TDM_SIX_ARGS 0
# endif
#else
# define TDM_SIX_ARGS 0
#endif

#if HAVE_TDM
// 2-D tile DMA: global (row-major, row stride d0_stride elems of 2B) -> LDS,
// writing tile_d0-elem rows with LDS padding (pad_iv: 2^(v+1) DWORDs between
// pads, pad_amt: v+1 DWORDs of pad) so the tile lands in the bank-padded
// layout the compute side indexes.  One issue per calling wave (EXEC ignored).
__device__ __forceinline__ void tdm_load_2d(unsigned lds_off, const void* gaddr,
                                            unsigned tile_d0, unsigned tile_d1,
                                            unsigned d0_stride,
                                            unsigned pad_iv, unsigned pad_amt)
{
    const unsigned long long ga = (unsigned long long)gaddr;
    u32x4 g0;
    g0[0] = 1u;                                            // count=1, user mode
    g0[1] = lds_off;                                       // LDS byte address
    g0[2] = (unsigned)ga;                                  // global_addr[31:0]
    g0[3] = (unsigned)((ga >> 32) & 0x1FFFFFFu) | (2u << 30); // addr[56:32], type=2
    i32x8 g1;
    g1[0] = (int)((1u << 16) | (1u << 20) | (pad_iv << 22) | (pad_amt << 25)); // 2B elems, pad_enable
    g1[1] = (int)((tile_d0 & 0xFFFFu) << 16);              // tensor_dim0 = tile_d0 (in-bounds tile)
    g1[2] = (int)((tile_d0 >> 16) | ((tile_d1 & 0xFFFFu) << 16)); // tensor_dim1 = tile_d1
    g1[3] = (int)((tile_d1 >> 16) | (tile_d0 << 16));      // tile_dim0
    g1[4] = (int)(tile_d1 & 0xFFFFu);                      // tile_dim1, tile_dim2=0
    g1[5] = (int)d0_stride;                                // tensor_dim0_stride[31:0]
    g1[6] = 0;                                             // stride[47:32]=0, dim1_stride lo=0
    g1[7] = 0;
    const i32x4 z4 = {0, 0, 0, 0};
#if TDM_SIX_ARGS
    const i32x8 z8 = {0, 0, 0, 0, 0, 0, 0, 0};
    __builtin_amdgcn_tensor_load_to_lds(g0, g1, z4, z4, z8, 0);
#else
    __builtin_amdgcn_tensor_load_to_lds(g0, g1, z4, z4, 0);
#endif
}
#endif

// ---------------------------------------------------------------- RoI align
__global__ __launch_bounds__(256)
void roi_align_kernel(const float* __restrict__ prop, const int* __restrict__ imidx,
                      const float* __restrict__ f0, const float* __restrict__ f1,
                      const float* __restrict__ f2, const float* __restrict__ f3,
                      bf16t* __restrict__ feats)
{
    const int p = blockIdx.x;
    const int c = threadIdx.x;              // lane = channel 0..255
    __shared__ int   sx0[14], sx1[14], sy0[14], sy1[14], sex[14], sey[14];
    __shared__ float slx[14], shx[14], sly[14], shy[14];

    const float px1 = prop[p*4+0], py1 = prop[p*4+1];
    const float px2 = prop[p*4+2], py2 = prop[p*4+3];
    const float wb = px2 - px1, hb = py2 - py1;
    float lv = floorf(4.0f + log2f(sqrtf(wb*hb) / 224.0f + 1e-8f));
    lv = fminf(fmaxf(lv, 2.0f), 5.0f);
    const int li = (int)lv - 2;
    const float* fm; int H, W; float inv;
    if      (li == 0) { fm = f0; H = 200; W = 200; inv = 0.25f;    }
    else if (li == 1) { fm = f1; H = 100; W = 100; inv = 0.125f;   }
    else if (li == 2) { fm = f2; H =  50; W =  50; inv = 0.0625f;  }
    else              { fm = f3; H =  25; W =  25; inv = 0.03125f; }
    const float x1 = px1*inv, y1 = py1*inv, x2 = px2*inv, y2 = py2*inv;
    const float bw = fmaxf(x2 - x1, 1.0f) * (1.0f/7.0f);
    const float bh = fmaxf(y2 - y1, 1.0f) * (1.0f/7.0f);

    if (c < 14) {
        const int s = c;
        const float g  = (float)(s >> 1) + ((float)(s & 1) + 0.5f) * 0.5f;
        const float xs = x1 + g * bw;
        sex[s] = (xs < -1.0f) || (xs > (float)W);
        const float xc = fminf(fmaxf(xs, 0.0f), (float)(W - 1));
        const int xi0 = (int)floorf(xc);
        sx0[s] = xi0;
        sx1[s] = (xi0 + 1 < W - 1) ? xi0 + 1 : W - 1;
        const float lx = xc - (float)xi0;
        slx[s] = lx; shx[s] = 1.0f - lx;
    } else if (c < 28) {
        const int s = c - 14;
        const float g  = (float)(s >> 1) + ((float)(s & 1) + 0.5f) * 0.5f;
        const float ys = y1 + g * bh;
        sey[s] = (ys < -1.0f) || (ys > (float)H);
        const float yc = fminf(fmaxf(ys, 0.0f), (float)(H - 1));
        const int yi0 = (int)floorf(yc);
        sy0[s] = yi0;
        sy1[s] = (yi0 + 1 < H - 1) ? yi0 + 1 : H - 1;
        const float ly = yc - (float)yi0;
        sly[s] = ly; shy[s] = 1.0f - ly;
    }
    __syncthreads();

    const int b = imidx[p];
    const float* base = fm + (size_t)(b * C_CH + c) * H * W;
    bf16t* out = feats + (size_t)p * K1 + c * 49;
    for (int iy = 0; iy < 7; ++iy) {
        for (int ix = 0; ix < 7; ++ix) {
            float acc = 0.0f;
            #pragma unroll
            for (int a = 0; a < 2; ++a) {
                const int sy = 2*iy + a;
                if (sey[sy]) continue;                      // uniform across wave
                const float* r0 = base + sy0[sy] * W;
                const float* r1 = base + sy1[sy] * W;
                const float ly = sly[sy], hy = shy[sy];
                #pragma unroll
                for (int q = 0; q < 2; ++q) {
                    const int sx = 2*ix + q;
                    if (sex[sx]) continue;
                    const int xa = sx0[sx], xb = sx1[sx];
                    const float lx = slx[sx], hx = shx[sx];
                    acc += hy*(hx*r0[xa] + lx*r0[xb]) + ly*(hx*r1[xa] + lx*r1[xb]);
                }
            }
            out[iy*7 + ix] = (bf16t)(acc * 0.25f);
        }
    }
}

// ------------------------------------------------------------- helpers
__global__ void zero_pad_kernel(bf16t* __restrict__ feats)
{
    const int i = blockIdx.x * 256 + threadIdx.x;
    if (i < (MROWS - N_PROP) * K1)
        feats[(size_t)N_PROP * K1 + i] = (bf16t)0.0f;
}

__global__ void cvt_bf16_kernel(const float* __restrict__ src, bf16t* __restrict__ dst, int n)
{
    const int i = blockIdx.x * 256 + threadIdx.x;
    if (i < n) dst[i] = (bf16t)src[i];
}

// ------------------------------------------------------------- WMMA GEMM
// C[M=1024][N=1024] = relu(A[M][K] * B[K][N] + bias[N]); A,B bf16 row-major.
// Block tile 128x64, wave tile 32x32 (4 wmma/wave/chunk), TDM double-buffered.
template<bool OUT_BF16>
__global__ __launch_bounds__(256)
void gemm_kernel(const bf16t* __restrict__ A, const bf16t* __restrict__ B,
                 const float* __restrict__ bias, void* __restrict__ Cout, int K)
{
    __shared__ __align__(16) bf16t lA[2][128 * ASTR];   // 2 x 10240 B
    __shared__ __align__(16) bf16t lB[2][32 * BSTR];    // 2 x  4608 B
    const int tid  = threadIdx.x;
    const int lane = tid & 31;
    const int wave = tid >> 5;                  // 0..7
    const int bm = blockIdx.x * 128;
    const int bn = blockIdx.y * 64;
    const int wm = (wave & 3) * 32;             // 4 waves along M
    const int wn = (wave >> 2) * 32;            // 2 waves along N
    v8f acc00 = {}, acc01 = {}, acc10 = {}, acc11 = {};

    const int akb = (lane >> 4) * 8;            // A-frag K half select
    const int am0 = wm + (lane & 15);
    const int am1 = am0 + 16;

    auto compute = [&](const bf16t* bufA, const bf16t* bufB) {
        v8bf a0lo = *(const v8bf*)&bufA[am0 * ASTR + akb];
        v8bf a0hi = *(const v8bf*)&bufA[am0 * ASTR + akb + 16];
        v16bf af0 = __builtin_shufflevector(a0lo, a0hi, 0,1,2,3,4,5,6,7,8,9,10,11,12,13,14,15);
        v8bf a1lo = *(const v8bf*)&bufA[am1 * ASTR + akb];
        v8bf a1hi = *(const v8bf*)&bufA[am1 * ASTR + akb + 16];
        v16bf af1 = __builtin_shufflevector(a1lo, a1hi, 0,1,2,3,4,5,6,7,8,9,10,11,12,13,14,15);
        const bf16t* brp = &bufB[akb * BSTR + wn + (lane & 15)];     // B-frag: lane&15 = n col, k table order
        v16bf bf0, bf1;
        #pragma unroll
        for (int s = 0; s < 8; ++s) {
            bf0[s]     = brp[s * BSTR];
            bf0[s + 8] = brp[(16 + s) * BSTR];
            bf1[s]     = brp[s * BSTR + 16];
            bf1[s + 8] = brp[(16 + s) * BSTR + 16];
        }
        acc00 = __builtin_amdgcn_wmma_f32_16x16x32_bf16(false, af0, false, bf0, (short)0, acc00, false, false);
        acc01 = __builtin_amdgcn_wmma_f32_16x16x32_bf16(false, af0, false, bf1, (short)0, acc01, false, false);
        acc10 = __builtin_amdgcn_wmma_f32_16x16x32_bf16(false, af1, false, bf0, (short)0, acc10, false, false);
        acc11 = __builtin_amdgcn_wmma_f32_16x16x32_bf16(false, af1, false, bf1, (short)0, acc11, false, false);
    };

    const int nk = K / 32;
#if HAVE_TDM
    // ---- TDM double-buffered pipeline: wave0 issues async tile DMAs --------
    if (wave == 0) {
        tdm_load_2d((unsigned)(size_t)&lA[0][0], A + (size_t)bm * K,
                    32u, 128u, (unsigned)K, 3u /*16 DW*/, 3u /*4 DW pad*/);
        tdm_load_2d((unsigned)(size_t)&lB[0][0], B + (size_t)0 * CLIN + bn,
                    64u, 32u, (unsigned)CLIN, 4u /*32 DW*/, 3u /*4 DW pad*/);
        __builtin_amdgcn_s_wait_tensorcnt(0);
    }
    __syncthreads();
    for (int kc = 0; kc < nk; ++kc) {
        const int cur = kc & 1;
        if (wave == 0 && (kc + 1) < nk) {
            const int k0 = (kc + 1) * 32;
            tdm_load_2d((unsigned)(size_t)&lA[cur ^ 1][0], A + (size_t)bm * K + k0,
                        32u, 128u, (unsigned)K, 3u, 3u);
            tdm_load_2d((unsigned)(size_t)&lB[cur ^ 1][0], B + (size_t)k0 * CLIN + bn,
                        64u, 32u, (unsigned)CLIN, 4u, 3u);
        }
        compute(&lA[cur][0], &lB[cur][0]);      // overlaps with the DMA
        if (wave == 0) __builtin_amdgcn_s_wait_tensorcnt(0);
        __syncthreads();
    }
#else
    // ---- fallback: cooperative synchronous copy ----------------------------
    for (int kc = 0; kc < nk; ++kc) {
        __syncthreads();
        const int k0 = kc * 32;
        const int r = tid >> 2, cc = (tid & 3) * 8;       // A 128x32: 2 rows/thread
        *(v8bf*)&lA[0][r * ASTR + cc]        = *(const v8bf*)&A[(size_t)(bm + r) * K + k0 + cc];
        *(v8bf*)&lA[0][(r + 64) * ASTR + cc] = *(const v8bf*)&A[(size_t)(bm + r + 64) * K + k0 + cc];
        const int br = tid >> 3, bc = (tid & 7) * 8;      // B 32x64
        *(v8bf*)&lB[0][br * BSTR + bc] = *(const v8bf*)&B[(size_t)(k0 + br) * CLIN + bn + bc];
        __syncthreads();
        compute(&lA[0][0], &lB[0][0]);
    }
#endif

    // ---- epilogue: C layout VGPR v -> m = 8*(lane>>4)+v, n = lane&15 -------
    const int cn  = lane & 15;
    const int cm0 = (lane >> 4) * 8;
    auto store_tile = [&](const v8f& acc, int mt, int nt) {
        #pragma unroll
        for (int v = 0; v < 8; ++v) {
            const int m = bm + wm + mt + cm0 + v;
            const int n = bn + wn + nt + cn;
            const float val = fmaxf(acc[v] + bias[n], 0.0f);
            if constexpr (OUT_BF16) ((bf16t*)Cout)[(size_t)m * CLIN + n] = (bf16t)val;
            else                    ((float*)Cout)[(size_t)m * CLIN + n] = val;
        }
    };
    store_tile(acc00,  0,  0);
    store_tile(acc01,  0, 16);
    store_tile(acc10, 16,  0);
    store_tile(acc11, 16, 16);
}

// -------------------------------------------------- head + box decode (nc == 1)
__global__ __launch_bounds__(256)
void head_decode_kernel(const float* __restrict__ h2,
                        const float* __restrict__ wc, const float* __restrict__ bc,
                        const float* __restrict__ wr, const float* __restrict__ br,
                        const float* __restrict__ prop, const int* __restrict__ imidx,
                        const float* __restrict__ imsz,
                        float* __restrict__ oboxes, float* __restrict__ oscores)
{
    const int r = blockIdx.x;
    const int t = threadIdx.x;
    const float* x = h2 + (size_t)r * CLIN;
    float a0=0,a1=0,a2=0,a3=0,a4=0,a5=0;
    for (int k = t; k < CLIN; k += 256) {
        const float xv = x[k];
        a0 += xv * wc[k*2+0]; a1 += xv * wc[k*2+1];
        a2 += xv * wr[k*4+0]; a3 += xv * wr[k*4+1];
        a4 += xv * wr[k*4+2]; a5 += xv * wr[k*4+3];
    }
    __shared__ float red[6][256];
    red[0][t]=a0; red[1][t]=a1; red[2][t]=a2; red[3][t]=a3; red[4][t]=a4; red[5][t]=a5;
    __syncthreads();
    for (int st = 128; st > 0; st >>= 1) {
        if (t < st) {
            #pragma unroll
            for (int q = 0; q < 6; ++q) red[q][t] += red[q][t + st];
        }
        __syncthreads();
    }
    if (t == 0) {
        const float l0 = red[0][0] + bc[0];
        const float l1 = red[1][0] + bc[1];
        const float mx = fmaxf(l0, l1);
        const float e0 = expf(l0 - mx), e1 = expf(l1 - mx);
        const float s = e0 / (e0 + e1);                 // softmax, fg = col 0
        const float rg0 = red[2][0] + br[0];
        const float rg1 = red[3][0] + br[1];
        const float rg2 = red[4][0] + br[2];
        const float rg3 = red[5][0] + br[3];
        const float px1 = prop[r*4+0], py1 = prop[r*4+1];
        const float px2 = prop[r*4+2], py2 = prop[r*4+3];
        const float cx = (px1 + px2) * 0.5f, cy = (py1 + py2) * 0.5f;
        const float pw = px2 - px1, ph = py2 - py1;
        const float MAXLOG = 4.135166556742356f;        // log(1000/16)
        const float dw = fminf(rg2 * 0.2f, MAXLOG);
        const float dh = fminf(rg3 * 0.2f, MAXLOG);
        const float ncx = cx + rg0 * 0.1f * pw;
        const float ncy = cy + rg1 * 0.1f * ph;
        const float nw = pw * expf(dw), nh = ph * expf(dh);
        const int im = imidx[r];
        const float sh = imsz[im*2+0], sw = imsz[im*2+1];
        const float bx1 = fminf(fmaxf(ncx - nw*0.5f, 0.0f), sw);
        const float bx2 = fminf(fmaxf(ncx + nw*0.5f, 0.0f), sw);
        const float by1 = fminf(fmaxf(ncy - nh*0.5f, 0.0f), sh);
        const float by2 = fminf(fmaxf(ncy + nh*0.5f, 0.0f), sh);
        oboxes[r*4+0]=bx1; oboxes[r*4+1]=by1; oboxes[r*4+2]=bx2; oboxes[r*4+3]=by2;
        const bool m = (s > 0.05f) && ((bx2-bx1) >= 1.0f) && ((by2-by1) >= 1.0f);
        oscores[r] = m ? s : -1.0f;
    }
}

// --------------------------------------------------------------- greedy NMS
__global__ __launch_bounds__(256)
void nms_kernel(const float* __restrict__ boxes, const float* __restrict__ scores,
                const int* __restrict__ imidx, float* __restrict__ out)
{
    const int img = blockIdx.x;
    const int t = threadIdx.x;
    __shared__ float s[N_PROP];
    __shared__ float bx[N_PROP * 4];
    __shared__ float area[N_PROP];
    __shared__ float r_val[256];
    __shared__ int   r_idx[256];
    __shared__ int   sel_i;
    __shared__ float sel_s;

    for (int i = t; i < N_PROP; i += 256) {
        s[i] = (imidx[i] == img) ? scores[i] : -1.0f;
        const float x1 = boxes[i*4+0], y1 = boxes[i*4+1];
        const float x2 = boxes[i*4+2], y2 = boxes[i*4+3];
        bx[i*4+0]=x1; bx[i*4+1]=y1; bx[i*4+2]=x2; bx[i*4+3]=y2;
        area[i] = (x2 - x1) * (y2 - y1);
    }
    __syncthreads();

    for (int it = 0; it < IMTOP; ++it) {
        float bv = -1e30f; int bi = 0;
        for (int i = t; i < N_PROP; i += 256) {
            const float v = s[i];
            if (v > bv) { bv = v; bi = i; }
        }
        r_val[t] = bv; r_idx[t] = bi;
        __syncthreads();
        for (int st = 128; st > 0; st >>= 1) {
            if (t < st) {
                const float ov = r_val[t + st]; const int oi = r_idx[t + st];
                if (ov > r_val[t] || (ov == r_val[t] && oi < r_idx[t])) {
                    r_val[t] = ov; r_idx[t] = oi;
                }
            }
            __syncthreads();
        }
        if (t == 0) { sel_i = r_idx[0]; sel_s = r_val[0]; }
        __syncthreads();
        const int   i  = sel_i;
        const float si = sel_s;
        const bool valid = si > 0.0f;
        if (t == 0) {
            const float vf = valid ? 1.0f : 0.0f;
            out[(img*IMTOP + it)*4 + 0] = bx[i*4+0] * vf;
            out[(img*IMTOP + it)*4 + 1] = bx[i*4+1] * vf;
            out[(img*IMTOP + it)*4 + 2] = bx[i*4+2] * vf;
            out[(img*IMTOP + it)*4 + 3] = bx[i*4+3] * vf;
            out[N_IMG*IMTOP*4 + img*IMTOP + it] = valid ? si : 0.0f;
            out[N_IMG*IMTOP*5 + img*IMTOP + it] = valid ? 0.0f : -1.0f;
        }
        const float xi1 = bx[i*4+0], yi1 = bx[i*4+1];
        const float xi2 = bx[i*4+2], yi2 = bx[i*4+3];
        const float ai  = area[i];
        __syncthreads();
        for (int j = t; j < N_PROP; j += 256) {
            const float xx1 = fmaxf(xi1, bx[j*4+0]);
            const float yy1 = fmaxf(yi1, bx[j*4+1]);
            const float xx2 = fminf(xi2, bx[j*4+2]);
            const float yy2 = fminf(yi2, bx[j*4+3]);
            const float inter = fmaxf(xx2 - xx1, 0.0f) * fmaxf(yy2 - yy1, 0.0f);
            const float iou = inter / (ai + area[j] - inter + 1e-9f);
            if (iou > 0.5f) s[j] = -1.0f;
        }
        __syncthreads();
    }
}

// ------------------------------------------------------------------ launch
extern "C" void kernel_launch(void* const* d_in, const int* in_sizes, int n_in,
                              void* d_out, int out_size, void* d_ws, size_t ws_size,
                              hipStream_t stream)
{
    const float* proposals = (const float*)d_in[0];
    const int*   imidx     = (const int*)  d_in[1];
    const float* fmap0     = (const float*)d_in[2];
    const float* fmap1     = (const float*)d_in[3];
    const float* fmap2     = (const float*)d_in[4];
    const float* fmap3     = (const float*)d_in[5];
    const float* imsizes   = (const float*)d_in[6];
    const float* w_fc1     = (const float*)d_in[7];
    const float* b_fc1     = (const float*)d_in[8];
    const float* w_fc2     = (const float*)d_in[9];
    const float* b_fc2     = (const float*)d_in[10];
    const float* w_cls     = (const float*)d_in[11];
    const float* b_cls     = (const float*)d_in[12];
    const float* w_reg     = (const float*)d_in[13];
    const float* b_reg     = (const float*)d_in[14];

    char* w = (char*)d_ws;
    bf16t* feats = (bf16t*)w; w += (size_t)MROWS * K1 * 2;     // 25.7 MB
    bf16t* wt1   = (bf16t*)w; w += (size_t)K1 * CLIN * 2;      // 25.7 MB
    bf16t* wt2   = (bf16t*)w; w += (size_t)CLIN * CLIN * 2;    //  2.1 MB
    bf16t* h1    = (bf16t*)w; w += (size_t)MROWS * CLIN * 2;   //  2.1 MB
    float* h2    = (float*)w; w += (size_t)MROWS * CLIN * 4;   //  4.2 MB
    float* dbox  = (float*)w; w += (size_t)N_PROP * 4 * 4;
    float* dscr  = (float*)w; w += (size_t)N_PROP * 4;

    const int n1 = K1 * CLIN;
    const int n2 = CLIN * CLIN;
    cvt_bf16_kernel<<<(n1 + 255) / 256, 256, 0, stream>>>(w_fc1, wt1, n1);
    cvt_bf16_kernel<<<(n2 + 255) / 256, 256, 0, stream>>>(w_fc2, wt2, n2);
    zero_pad_kernel<<<((MROWS - N_PROP) * K1 + 255) / 256, 256, 0, stream>>>(feats);
    roi_align_kernel<<<N_PROP, 256, 0, stream>>>(proposals, imidx, fmap0, fmap1, fmap2, fmap3, feats);
    gemm_kernel<true ><<<dim3(MROWS/128, CLIN/64), 256, 0, stream>>>(feats, wt1, b_fc1, (void*)h1, K1);
    gemm_kernel<false><<<dim3(MROWS/128, CLIN/64), 256, 0, stream>>>(h1,    wt2, b_fc2, (void*)h2, CLIN);
    head_decode_kernel<<<N_PROP, 256, 0, stream>>>(h2, w_cls, b_cls, w_reg, b_reg,
                                                   proposals, imidx, imsizes, dbox, dscr);
    nms_kernel<<<N_IMG, 256, 0, stream>>>(dbox, dscr, imidx, (float*)d_out);
}
